// SpatialEmbLoss_27204322853617
// MI455X (gfx1250) — matrix operand order repaired and checked by reading, not compiled, not run on hardware
//
#include <hip/hip_runtime.h>
#include <hip/hip_bf16.h>

typedef __attribute__((ext_vector_type(16))) _Float16 v16h;
typedef __attribute__((ext_vector_type(8)))  float    v8f;

namespace {
constexpr int kB = 4, kC = 24, kH = 512, kW = 512, kP = kH * kW;
constexpr int kNI = 32, kNC = 20, kBins = 2048;
constexpr float kInvStep = 1.0f / 511.0f;   // linspace(0,1,512) step
}

// ---------------------------------------------------------------- zero scratch
__global__ void k_zero(unsigned* __restrict__ p, long n) {
  long i = (long)blockIdx.x * blockDim.x + threadIdx.x;
  if (i < n) p[i] = 0u;
}

// ---------------------------------------------------------------- phase 1:
// spatial_emb = tanh(pred[0:2]) + xy  (stored),  seed_bg sum, per-instance class
__global__ void __launch_bounds__(256) k_prep(const float* __restrict__ pred,
                                              const int* __restrict__ inst,
                                              const int* __restrict__ lab,
                                              float* __restrict__ emb,
                                              float* __restrict__ accSeedBg,
                                              int* __restrict__ clsmax) {
  __shared__ int lmax[kNI + 1];
  int tid = threadIdx.x;
  if (tid <= kNI) lmax[tid] = 0;
  __syncthreads();
  long lin = (long)blockIdx.x * blockDim.x + tid;   // kP % 256 == 0 -> block never crosses images
  int b = (int)(lin / kP);
  int p = (int)(lin % kP);
  const float* pb = pred + (size_t)b * kC * kP;
  float x = (float)(p & (kW - 1)) * kInvStep;
  float y = (float)((p >> 9) & (kH - 1)) * kInvStep;
  emb[((size_t)b * 2 + 0) * kP + p] = tanhf(pb[(size_t)0 * kP + p]) + x;
  emb[((size_t)b * 2 + 1) * kP + p] = tanhf(pb[(size_t)1 * kP + p]) + y;
  int lv = lab[(size_t)b * kP + p];
  // seed_bg = sum_c seed_c^2 * (lab != c+1); valid==true always (labels in 0..20)
  float ss = 0.f, sub = 0.f;
  #pragma unroll
  for (int c = 0; c < kNC; ++c) {
    float s = 1.f / (1.f + expf(-pb[(size_t)(4 + c) * kP + p]));
    float s2 = s * s;
    ss += s2;
    if (lv == c + 1) sub = s2;
  }
  float a = ss - sub;
  for (int off = 16; off > 0; off >>= 1) a += __shfl_down(a, off, 32);
  if ((tid & 31) == 0) atomicAdd(&accSeedBg[b], a);
  int iv = inst[(size_t)b * kP + p];
  if (iv > 0 && iv <= kNI) atomicMax(&lmax[iv], lv);
  __syncthreads();
  if (tid < kNI && lmax[tid + 1] > 0) atomicMax(&clsmax[b * kNI + tid], lmax[tid + 1]);
}

// ---------------------------------------------------------------- phase 2:
// per-instance masked feature sums as WMMA GEMM:
//   D[feat][inst] += A(16x32 features, f16) x B(32x16 one-hot, f16), f32 accum.
// A layout (ISA 7.12.2, 16-bit A 16x32): lane supplies row M=lane%16; half h ->
//   K = h<8 ? (lane<16?0:8)+h : (lane<16?16:24)+h-8.
// B layout (16-bit B 32x16): lane supplies col N=lane%16; half h -> same K as A (Kb = K).
// C/D (f32 16x16): lanes 0-15 hold M=r (r=VGPR idx) for N=lane.
// Feature row selection is done with lane-constant {0,1} weights -> branch-free
// FMA chain, EXEC stays all-ones through the whole WMMA loop (no saveexec nests).
__global__ void __launch_bounds__(128) k_stats(const float* __restrict__ pred,
                                               const int* __restrict__ inst,
                                               float* __restrict__ stats) {
  const int lane = threadIdx.x & 31;
  const int wv = threadIdx.x >> 5;
  const int blkPerImg = 64;
  const int b = blockIdx.x / blkPerImg;
  const int waveInImg = (blockIdx.x % blkPerImg) * 4 + wv;
  const int groupsPerWave = (kP / 32) / (blkPerImg * 4);   // 32
  const float* s0p = pred + ((size_t)b * kC + 2) * kP;
  const float* s1p = pred + ((size_t)b * kC + 3) * kP;
  const int* ip = inst + (size_t)b * kP;
  v8f acc0 = {}; v8f acc1 = {};
  const int f = lane & 15;      // feature row this lane supplies to A
  const int ncol = lane & 15;   // instance column this lane supplies to B
  // lane-constant feature weights (hoisted out of the loop; no divergence inside)
  const float w1  = (f == 0) ? 1.f : 0.f;   // ones -> count
  const float wx  = (f == 1) ? 1.f : 0.f;   // x
  const float wy  = (f == 2) ? 1.f : 0.f;   // y
  const float ws0 = (f == 3) ? 1.f : 0.f;   // sigma0
  const float ws1 = (f == 4) ? 1.f : 0.f;   // sigma1
  const float wq0 = (f == 5) ? 1.f : 0.f;   // sigma0^2
  const float wq1 = (f == 6) ? 1.f : 0.f;   // sigma1^2
  for (int gi = 0; gi < groupsPerWave; ++gi) {
    int p0 = (waveInImg * groupsPerWave + gi) * 32;
    int p = p0 + lane;
    int iv = ip[p];             // coalesced: lane k owns pixel p0+k
    float s0v = s0p[p];
    float s1v = s1p[p];
    v16h a, bm0, bm1;
    #pragma unroll
    for (int h = 0; h < 16; ++h) {
      int K = (h < 8) ? ((lane < 16 ? 0 : 8) + h) : ((lane < 16 ? 16 : 24) + h - 8);
      int pk = p0 + K;
      float s0k = __shfl(s0v, K, 32);   // ds_bpermute gather of pixel K's sigma
      float s1k = __shfl(s1v, K, 32);
      float xk = (float)(pk & (kW - 1)) * kInvStep;
      float yk = (float)((pk >> 9) & (kH - 1)) * kInvStep;
      float v = w1;
      v = fmaf(wx, xk, v);
      v = fmaf(wy, yk, v);
      v = fmaf(ws0, s0k, v);
      v = fmaf(ws1, s1k, v);
      v = fmaf(wq0, s0k * s0k, v);
      v = fmaf(wq1, s1k * s1k, v);
      a[h] = (_Float16)v;
      int Kb = K;
      int ik = __shfl(iv, Kb, 32);
      bm0[h] = (_Float16)((ik == ncol + 1) ? 1.f : 0.f);        // instances 1..16
      bm1[h] = (_Float16)((ik == ncol + 17) ? 1.f : 0.f);       // instances 17..32
    }
    acc0 = __builtin_amdgcn_wmma_f32_16x16x32_f16(false, a, false, bm0, (short)0, acc0, false, false);
    acc1 = __builtin_amdgcn_wmma_f32_16x16x32_f16(false, a, false, bm1, (short)0, acc1, false, false);
  }
  if (lane < 16) {   // feature rows 0..6 live in VGPRs 0..6 of lanes 0-15
    #pragma unroll
    for (int r = 0; r < 7; ++r) {
      atomicAdd(&stats[((size_t)b * kNI + ncol) * 8 + r], acc0[r]);
      atomicAdd(&stats[((size_t)b * kNI + 16 + ncol) * 8 + r], acc1[r]);
    }
  }
}

// ---------------------------------------------------------------- phase 3: finalize stats
__global__ void k_final_stats(const float* __restrict__ stats, const int* __restrict__ clsmax,
                              float* __restrict__ params, int* __restrict__ clsarr,
                              float* __restrict__ accVar) {
  int idx = threadIdx.x;
  if (idx >= kB * kNI) return;
  int b = idx / kNI;
  const float* s = stats + (size_t)idx * 8;
  float sumM = s[0];                 // gts = count of mask pixels
  float cnt = fmaxf(sumM, 1.f);
  float cx = s[1] / cnt, cy = s[2] / cnt;
  float m0 = s[3] / cnt, m1 = s[4] / cnt;
  float var0 = s[5] - 2.f * m0 * s[3] + m0 * m0 * sumM;   // sum (sig0-mean)^2 * m
  float var1 = s[6] - 2.f * m1 * s[4] + m1 * m1 * sumM;
  atomicAdd(&accVar[b], (var0 + var1) / (2.f * cnt));
  float* pp = params + (size_t)idx * 8;
  pp[0] = cx; pp[1] = cy;
  pp[2] = expf(10.f * m0); pp[3] = expf(10.f * m1);
  pp[4] = sumM;
  int cls = clsmax[idx] - 1;
  clsarr[idx] = min(max(cls, 0), kNC - 1);   // jnp.take clips
}

// ---------------------------------------------------------------- phase 4:
// per (image,instance): dist, seed_fg, and the error histogram (replaces top_k sort)
__global__ void __launch_bounds__(256) k_dist(const float* __restrict__ pred,
        const int* __restrict__ inst, const float* __restrict__ emb,
        const float* __restrict__ params, const int* __restrict__ clsarr,
        unsigned* __restrict__ hfg, unsigned* __restrict__ hbg, float* __restrict__ hse,
        float* __restrict__ accSeedFg) {
  __shared__ unsigned sfg[kBins];
  __shared__ unsigned sbg[kBins];
  __shared__ float sse[kBins];
  const int tid = threadIdx.x;
  for (int j = tid; j < kBins; j += 256) { sfg[j] = 0u; sbg[j] = 0u; sse[j] = 0.f; }
  const int b = blockIdx.z, ii = blockIdx.y;
  const int iid = ii + 1;
  const int idx = b * kNI + ii;
  const float cx = params[idx * 8 + 0], cy = params[idx * 8 + 1];
  const float sA = params[idx * 8 + 2], sB = params[idx * 8 + 3];
  const int cls = clsarr[idx];
  const float* ex_p = emb + (size_t)b * 2 * kP;
  const float* ey_p = ex_p + kP;
  const int* ip = inst + (size_t)b * kP;
  const float* sl_p = pred + ((size_t)b * kC + 4 + cls) * kP;
  __syncthreads();
  float sf = 0.f;
  const int chunk = kP / gridDim.x;        // 32768 pixels per block
  const int base = blockIdx.x * chunk;
  for (int k = 0; k < chunk / 256; ++k) {
    int p = base + k * 256 + tid;
    __builtin_prefetch(&sl_p[p + 2048], 0, 1);   // -> global_prefetch_b8
    __builtin_prefetch(&ex_p[p + 2048], 0, 1);
    float ex = ex_p[p], ey = ey_p[p];
    int iv = ip[p];
    float sig = 1.f / (1.f + expf(-sl_p[p]));
    float dx = ex - cx, dy = ey - cy;
    float dist = expf(-(dx * dx * sA + dy * dy * sB));
    bool g = (iv == iid);
    float e = g ? 2.f * (1.f - dist) : 2.f * dist;   // hinge error, in [0,2]
    int bin = (int)(e * (kBins * 0.5f));
    bin = bin < 0 ? 0 : (bin > kBins - 1 ? kBins - 1 : bin);
    atomicAdd(&sse[bin], e);
    atomicAdd(g ? &sfg[bin] : &sbg[bin], 1u);
    if (g) { float d = sig - dist; sf += d * d; }
  }
  for (int off = 16; off > 0; off >>= 1) sf += __shfl_down(sf, off, 32);
  if ((tid & 31) == 0) atomicAdd(&accSeedFg[b], 10.0f * sf);   // FG_WEIGHT
  __syncthreads();
  unsigned* gfg = hfg + (size_t)idx * kBins;
  unsigned* gbg = hbg + (size_t)idx * kBins;
  float* gse = hse + (size_t)idx * kBins;
  for (int j = tid; j < kBins; j += 256) {
    unsigned cf = sfg[j], cb = sbg[j];
    if (cf | cb) {
      if (cf) atomicAdd(&gfg[j], cf);
      if (cb) atomicAdd(&gbg[j], cb);
      atomicAdd(&gse[j], sse[j]);
    }
  }
}

// ---------------------------------------------------------------- phase 5:
// Lovasz hinge via descending-bin scan. sum_i e_i*(jac_i - jac_{i-1}) telescopes
// per bin to mean_e * (J(counts incl bin) - J(counts before bin)).
__global__ void __launch_bounds__(32) k_lovasz(const unsigned* __restrict__ hfg,
        const unsigned* __restrict__ hbg, const float* __restrict__ hse,
        const float* __restrict__ params, float* __restrict__ accInst) {
  const int idx = blockIdx.x;           // b*32 + inst
  const int b = idx / kNI;
  const int lane = threadIdx.x;
  const float gts = params[idx * 8 + 4];
  const unsigned* fgp = hfg + (size_t)idx * kBins;
  const unsigned* bgp = hbg + (size_t)idx * kBins;
  const float* sep = hse + (size_t)idx * kBins;
  float cumf = 0.f, cumb = 0.f, loss = 0.f;
  for (int top = kBins - 32; top >= 0; top -= 32) {
    int bin = top + (31 - lane);        // lane 0 = largest error bin of this chunk
    float cf = (float)fgp[bin];
    float cb = (float)bgp[bin];
    float se = sep[bin];
    float inf_ = cf, inb_ = cb;         // inclusive prefix over lanes (descending bins)
    for (int d = 1; d < 32; d <<= 1) {
      float tf = __shfl_up(inf_, d, 32);
      float tb = __shfl_up(inb_, d, 32);
      if (lane >= d) { inf_ += tf; inb_ += tb; }
    }
    float cnt = cf + cb;
    if (cnt > 0.f) {
      float fg1 = cumf + inf_, bg1 = cumb + inb_;
      float fg0 = fg1 - cf,  bg0 = bg1 - cb;
      float J0 = (fg0 == 0.f && bg0 == 0.f) ? 0.f
               : 1.f - (gts - fg0) / fmaxf(gts + bg0, 1e-9f);
      float J1 = 1.f - (gts - fg1) / fmaxf(gts + bg1, 1e-9f);
      loss += (se / cnt) * (J1 - J0);
    }
    cumf += __shfl(inf_, 31, 32);
    cumb += __shfl(inb_, 31, 32);
  }
  for (int off = 16; off > 0; off >>= 1) loss += __shfl_down(loss, off, 32);
  if (lane == 0) atomicAdd(&accInst[b], loss);
}

// ---------------------------------------------------------------- phase 6: combine
__global__ void k_combine(const float* __restrict__ acc, float* __restrict__ out) {
  int lane = threadIdx.x;
  float t = 0.f;
  if (lane < kB) {
    float inst_l = acc[lane] / (float)kNI;
    float var_l  = acc[4 + lane] / (float)kNI;
    float seed_l = (acc[8 + lane] + acc[12 + lane]) / (float)(kH * kW);
    t = 1.0f * inst_l + 10.0f * var_l + 1.0f * seed_l;   // W_INST, W_VAR, W_SEED
  }
  for (int off = 16; off > 0; off >>= 1) t += __shfl_down(t, off, 32);
  if (lane == 0) out[0] = t / (float)kB;
}

extern "C" void kernel_launch(void* const* d_in, const int* in_sizes, int n_in,
                              void* d_out, int out_size, void* d_ws, size_t ws_size,
                              hipStream_t stream) {
  const float* pred = (const float*)d_in[0];
  const int* inst = (const int*)d_in[1];
  const int* lab = (const int*)d_in[2];
  float* out = (float*)d_out;

  // workspace layout (~11.5 MB)
  float* emb    = (float*)d_ws;                             // B*2*P f32
  float* stats  = emb + (size_t)kB * 2 * kP;                // B*32*8 (wmma sums)
  float* params = stats + kB * kNI * 8;                     // B*32*8 (cx,cy,sA,sB,gts)
  int* clsmax   = (int*)(params + kB * kNI * 8);            // B*32
  int* clsarr   = clsmax + kB * kNI;                        // B*32
  unsigned* hfg = (unsigned*)(clsarr + kB * kNI);           // B*32*kBins
  unsigned* hbg = hfg + (size_t)kB * kNI * kBins;
  float* hse    = (float*)(hbg + (size_t)kB * kNI * kBins);
  float* acc    = hse + (size_t)kB * kNI * kBins;           // [0:4) inst, [4:8) var,
                                                            // [8:12) seed_bg, [12:16) seed_fg
  unsigned* z0 = (unsigned*)stats;                          // zero stats..acc (contiguous)
  long zn = (long)((unsigned*)(acc + 16) - z0);

  k_zero<<<(unsigned)((zn + 255) / 256), 256, 0, stream>>>(z0, zn);
  k_prep<<<kB * kP / 256, 256, 0, stream>>>(pred, inst, lab, emb, acc + 8, clsmax);
  k_stats<<<kB * 64, 128, 0, stream>>>(pred, inst, stats);
  k_final_stats<<<1, 128, 0, stream>>>(stats, clsmax, params, clsarr, acc + 4);
  dim3 g4(8, kNI, kB);
  k_dist<<<g4, 256, 0, stream>>>(pred, inst, emb, params, clsarr, hfg, hbg, hse, acc + 12);
  k_lovasz<<<kB * kNI, 32, 0, stream>>>(hfg, hbg, hse, params, acc + 0);
  k_combine<<<1, 32, 0, stream>>>(acc, out);
}
